// AttentionDecoder_3289944949156
// MI455X (gfx1250) — matrix operand order repaired and checked by reading, not compiled, not run on hardware
//
#include <hip/hip_runtime.h>
#include <stdint.h>

#define HDIM 256
#define BDIM 64
#define SDIM 48
#define TDIM 48
#define VDIM 32000

typedef __attribute__((ext_vector_type(16))) __bf16 v16bf;
typedef __attribute__((ext_vector_type(8)))  __bf16 v8bf;
typedef __attribute__((ext_vector_type(8)))  float  v8f;

// ---------- helpers ----------
__device__ __forceinline__ unsigned short f2bf(float f) {
  union { float f; unsigned u; } v; v.f = f;
  unsigned u = v.u;
  if ((u & 0x7fffffffu) > 0x7f800000u) return (unsigned short)((u >> 16) | 0x0040u); // NaN
  u += 0x7fffu + ((u >> 16) & 1u); // round-to-nearest-even
  return (unsigned short)(u >> 16);
}

// A fragment out of LDS: two ds_load_b128 with offsets off one base.
__device__ __forceinline__ v16bf lds_afrag(const unsigned short* p, int off) {
  union { v16bf v; v8bf h[2]; } u;
  u.h[0] = *reinterpret_cast<const v8bf*>(p + off);
  u.h[1] = *reinterpret_cast<const v8bf*>(p + off + 8);
  return u.v;
}

__device__ __forceinline__ void store_tile(float* __restrict__ C, long long ldc,
                                           int rowBase, int col, const v8f& a) {
#pragma unroll
  for (int i = 0; i < 8; ++i) C[(long long)(rowBase + i) * ldc + col] = a[i];
}

#define WMMA_BF16(A, B, CACC)                                                  \
  __builtin_amdgcn_wmma_f32_16x16x32_bf16(false, (A), false, (B), (short)0,    \
                                          (CACC), false, false)

// ---------- setup kernels ----------
__global__ void cvt_f32_to_bf16(const float* __restrict__ in,
                                unsigned short* __restrict__ out, int n) {
  int i = blockIdx.x * blockDim.x + threadIdx.x;
  if (i < n) out[i] = f2bf(in[i]);
}

__global__ void init_h_kernel(const float* __restrict__ eh, float* __restrict__ h,
                              unsigned short* __restrict__ hB, int n) {
  int i = blockIdx.x * blockDim.x + threadIdx.x;
  if (i < n) { float v = eh[i]; h[i] = v; hB[i] = f2bf(v); }
}

// embB layout: [t][b][j] bf16, tokens = [SOS=0, target[:, :-1]]
__global__ void embed_kernel(const int* __restrict__ target,
                             const float* __restrict__ emb,
                             unsigned short* __restrict__ embB) {
  int i = blockIdx.x * blockDim.x + threadIdx.x;
  if (i >= TDIM * BDIM * HDIM) return;
  int j = i & (HDIM - 1);
  int b = (i >> 8) & 63;
  int t = i >> 14;
  int tok = (t == 0) ? 0 : target[b * TDIM + t - 1];
  embB[i] = f2bf(emb[(long long)tok * HDIM + j]);
}

// ---------- bf16 GEMM: C[M,N] = A[M,K] @ W[N,K]^T + bias[N] (f32 accumulate) ----
// lda == ldw == K at all call sites. One wave computes 64x32 of C (4 M-tiles x
// 2 N-tiles). The block's 8 waves share one 64-row A panel (nPairs % 8 == 0 at
// every call site -> mg is block-uniform), staged into padded LDS once; A
// fragments come from ds_load_b128 off a single 32-bit lane base. B uses two
// never-advanced global pointers. The k-loop is a REAL loop (unroll 1) with a
// 2-deep ping-pong so the scheduler cannot run ahead and spill accumulators.
// First k-step uses an inline-zero C; bias is added in the epilogue.
// Launch geometry is exact: total waves == gridDim.x * 8.
template <int K>
__global__ void gemm_bf16_kernel(const unsigned short* __restrict__ A,
                                 const unsigned short* __restrict__ W,
                                 const float* __restrict__ bias,
                                 float* __restrict__ C, long long ldc,
                                 int nPairs) {
  constexpr int LSTR = K + 8;               // padded row stride (elements):
                                            // keeps 16B align, breaks bank conflicts
  __shared__ unsigned short As[64 * LSTR];

  int tid = threadIdx.x;
  int lane = tid & 31;
  int wave = blockIdx.x * 8 + (tid >> 5);
  int np = wave % nPairs, mg = wave / nPairs;  // mg uniform across the block
  int m0 = mg * 64, n0 = np * 32;
  int nl = lane & 15, hi = lane >> 4;

  // ---- cooperative stage of A[m0..m0+63][0..K) into padded LDS ----
  {
    const unsigned short* Ag = A + (size_t)m0 * K;
    for (int c = tid * 8; c < 64 * K; c += 256 * 8) {
      int row = c / K, col = c % K;        // K is a power of two
      *reinterpret_cast<v8bf*>(&As[row * LSTR + col]) =
          *reinterpret_cast<const v8bf*>(Ag + c);
    }
  }
  __syncthreads();

  // Single 32-bit per-lane LDS base; row-block offsets are DS immediates.
  const unsigned short* Alds = &As[nl * LSTR + 16 * hi];
  // Two never-advanced global B pointers.
  const unsigned short* Wp0 = W + (size_t)(n0 + nl) * K + 16 * hi;
  const unsigned short* Wp1 = Wp0 + (size_t)16 * K;

  v8f acc0, acc1, acc2, acc3, acc4, acc5, acc6, acc7;
  v16bf a00, a01, a02, a03, b00, b01;   // buffer 0
  v16bf a10, a11, a12, a13, b10, b11;   // buffer 1

#define LOAD_SET(aA, aB, aC, aD, bA, bB, k0)                                   \
  do {                                                                         \
    bA = *reinterpret_cast<const v16bf*>(Wp0 + (k0));                          \
    bB = *reinterpret_cast<const v16bf*>(Wp1 + (k0));                          \
    aA = lds_afrag(Alds, 0 * 16 * LSTR + (k0));                                \
    aB = lds_afrag(Alds, 1 * 16 * LSTR + (k0));                                \
    aC = lds_afrag(Alds, 2 * 16 * LSTR + (k0));                                \
    aD = lds_afrag(Alds, 3 * 16 * LSTR + (k0));                                \
  } while (0)

#define MMA_SET(aA, aB, aC, aD, bA, bB)                                        \
  do {                                                                         \
    acc0 = WMMA_BF16(aA, bA, acc0);                                            \
    acc1 = WMMA_BF16(aA, bB, acc1);                                            \
    acc2 = WMMA_BF16(aB, bA, acc2);                                            \
    acc3 = WMMA_BF16(aB, bB, acc3);                                            \
    acc4 = WMMA_BF16(aC, bA, acc4);                                            \
    acc5 = WMMA_BF16(aC, bB, acc5);                                            \
    acc6 = WMMA_BF16(aD, bA, acc6);                                            \
    acc7 = WMMA_BF16(aD, bB, acc7);                                            \
  } while (0)

  // Preload k-steps 0 and 1; first MMA set uses inline-zero C.
  LOAD_SET(a00, a01, a02, a03, b00, b01, 0);
  LOAD_SET(a10, a11, a12, a13, b10, b11, 32);
  {
    const v8f zc = {0.f, 0.f, 0.f, 0.f, 0.f, 0.f, 0.f, 0.f};
    acc0 = WMMA_BF16(a00, b00, zc);
    acc1 = WMMA_BF16(a00, b01, zc);
    acc2 = WMMA_BF16(a01, b00, zc);
    acc3 = WMMA_BF16(a01, b01, zc);
    acc4 = WMMA_BF16(a02, b00, zc);
    acc5 = WMMA_BF16(a02, b01, zc);
    acc6 = WMMA_BF16(a03, b00, zc);
    acc7 = WMMA_BF16(a03, b01, zc);
  }
  // Steady state: {prefetch k, MMA k-1} in ping-pong pairs. K/32 is even
  // (8 or 16 steps); step 0 peeled above, last step finished after the loop.
#pragma unroll 1
  for (int k = 64; k < K; k += 64) {
    LOAD_SET(a00, a01, a02, a03, b00, b01, k);
    MMA_SET(a10, a11, a12, a13, b10, b11);
    LOAD_SET(a10, a11, a12, a13, b10, b11, k + 32);
    MMA_SET(a00, a01, a02, a03, b00, b01);
  }
  MMA_SET(a10, a11, a12, a13, b10, b11);

#undef LOAD_SET
#undef MMA_SET

  // Epilogue: bias add, then store.
  float bv0 = bias ? bias[n0 + nl] : 0.0f;
  float bv1 = bias ? bias[n0 + 16 + nl] : 0.0f;
#pragma unroll
  for (int i = 0; i < 8; ++i) {
    acc0[i] += bv0; acc2[i] += bv0; acc4[i] += bv0; acc6[i] += bv0;
    acc1[i] += bv1; acc3[i] += bv1; acc5[i] += bv1; acc7[i] += bv1;
  }

  // D layout: VGPR i -> row m0+16r+i+8*hi, col n0 + {0,16} + nl
  int rb = m0 + 8 * hi;
  store_tile(C, ldc, rb,      n0 + nl,      acc0);
  store_tile(C, ldc, rb,      n0 + 16 + nl, acc1);
  store_tile(C, ldc, rb + 16, n0 + nl,      acc2);
  store_tile(C, ldc, rb + 16, n0 + 16 + nl, acc3);
  store_tile(C, ldc, rb + 32, n0 + nl,      acc4);
  store_tile(C, ldc, rb + 32, n0 + 16 + nl, acc5);
  store_tile(C, ldc, rb + 48, n0 + nl,      acc6);
  store_tile(C, ldc, rb + 48, n0 + 16 + nl, acc7);
}

// ---------- attention (per batch row): scores, softmax, context, build x ----------
__global__ void attn_kernel(const float* __restrict__ q,      // [64,256] = h@Ww.T+Wb
                            const float* __restrict__ Ukeys,  // [64,48,256]
                            const float* __restrict__ Vw, const float* __restrict__ Vb,
                            const float* __restrict__ enc,    // [64,48,256] f32
                            const unsigned short* __restrict__ embB_t, // [64,256]
                            unsigned short* __restrict__ x,   // [64,512] bf16
                            float* __restrict__ attn_out) {   // base + t*SDIM
  int b = blockIdx.x, tid = threadIdx.x;
  __shared__ float ssh[SDIM];
  __shared__ float wsh[SDIM];
  __shared__ float ms[2];
  int wv = tid >> 5, ln = tid & 31;
  const float* qb = q + b * HDIM;

  for (int s = wv; s < SDIM; s += 8) {
    const float* uk = Ukeys + (long long)(b * SDIM + s) * HDIM;
    float p = 0.0f;
    for (int k = ln; k < HDIM; k += 32)
      p += Vw[k] * tanhf(qb[k] + uk[k]);
    for (int off = 16; off > 0; off >>= 1)
      p += __shfl_down(p, off, 32);
    if (ln == 0) ssh[s] = p + Vb[0];
  }
  __syncthreads();
  if (tid == 0) {
    float m = ssh[0];
    for (int s = 1; s < SDIM; ++s) m = fmaxf(m, ssh[s]);
    float sum = 0.0f;
    for (int s = 0; s < SDIM; ++s) sum += __expf(ssh[s] - m);
    ms[0] = m; ms[1] = sum;
  }
  __syncthreads();
  if (tid < SDIM) {
    float w = __expf(ssh[tid] - ms[0]) / ms[1];
    wsh[tid] = w;
    attn_out[(long long)b * (TDIM * SDIM) + tid] = w;  // attn[b][t][s]
  }
  __syncthreads();
  float c = 0.0f;
  const float* eb = enc + (long long)(b * SDIM) * HDIM + tid;
  for (int s = 0; s < SDIM; ++s) c += wsh[s] * eb[s * HDIM];
  x[b * (2 * HDIM) + HDIM + tid] = f2bf(c);          // ctx half
  x[b * (2 * HDIM) + tid] = embB_t[b * HDIM + tid];  // x_t half
}

// ---------- GRU elementwise ----------
__global__ void gru_kernel(const float* __restrict__ gi, const float* __restrict__ gh,
                           float* __restrict__ h, unsigned short* __restrict__ hB,
                           float* __restrict__ hlast) {
  int b = blockIdx.x, j = threadIdx.x;
  const float* gib = gi + b * 768;
  const float* ghb = gh + b * 768;
  float ir = gib[j], iz = gib[256 + j], in = gib[512 + j];
  float hr = ghb[j], hz = ghb[256 + j], hn = ghb[512 + j];
  float r = 1.0f / (1.0f + __expf(-(ir + hr)));
  float z = 1.0f / (1.0f + __expf(-(iz + hz)));
  float n = tanhf(in + r * hn);
  float ho = h[b * HDIM + j];
  float hnew = (1.0f - z) * n + z * ho;
  h[b * HDIM + j] = hnew;
  hB[b * HDIM + j] = f2bf(hnew);
  if (hlast) hlast[b * HDIM + j] = hnew;
}

// ---------- in-place log_softmax over V per row ----------
__global__ void logsoftmax_kernel(float* __restrict__ base, long long rowStride) {
  int b = blockIdx.x, tid = threadIdx.x;
  float* row = base + (long long)b * rowStride;
  __shared__ float red[256];
  float m = -3.4e38f;
  for (int i = tid; i < VDIM; i += 256) m = fmaxf(m, row[i]);
  red[tid] = m; __syncthreads();
  for (int s = 128; s > 0; s >>= 1) {
    if (tid < s) red[tid] = fmaxf(red[tid], red[tid + s]);
    __syncthreads();
  }
  m = red[0]; __syncthreads();
  float sum = 0.0f;
  for (int i = tid; i < VDIM; i += 256) sum += __expf(row[i] - m);
  red[tid] = sum; __syncthreads();
  for (int s = 128; s > 0; s >>= 1) {
    if (tid < s) red[tid] += red[tid + s];
    __syncthreads();
  }
  float ls = m + __logf(red[0]);
  for (int i = tid; i < VDIM; i += 256) row[i] = row[i] - ls;
}

// ---------- host ----------
extern "C" void kernel_launch(void* const* d_in, const int* in_sizes, int n_in,
                              void* d_out, int out_size, void* d_ws, size_t ws_size,
                              hipStream_t stream) {
  (void)in_sizes; (void)n_in; (void)out_size; (void)ws_size;
  const float* enc    = (const float*)d_in[0];
  const float* ehid   = (const float*)d_in[1];
  const int*   target = (const int*)d_in[2];
  const float* emb    = (const float*)d_in[3];
  const float* Ww     = (const float*)d_in[4];
  const float* Wb     = (const float*)d_in[5];
  const float* Uw     = (const float*)d_in[6];
  const float* Ub     = (const float*)d_in[7];
  const float* Vw     = (const float*)d_in[8];
  const float* Vb     = (const float*)d_in[9];
  const float* W_ih   = (const float*)d_in[10];
  const float* W_hh   = (const float*)d_in[11];
  const float* b_ih   = (const float*)d_in[12];
  const float* b_hh   = (const float*)d_in[13];
  const float* fc_w   = (const float*)d_in[14];
  const float* fc_b   = (const float*)d_in[15];

  float* out_logp  = (float*)d_out;                                // [64,48,32000]
  float* out_hlast = out_logp + (long long)BDIM * TDIM * VDIM;     // [1,64,256]
  float* out_attn  = out_hlast + BDIM * HDIM;                      // [64,48,48]

  char* ws = (char*)d_ws;
  size_t off = 0;
  auto alloc = [&](size_t bytes) -> char* {
    char* p = ws + off;
    off = (off + bytes + 255) & ~(size_t)255;
    return p;
  };
  unsigned short* encB = (unsigned short*)alloc((size_t)BDIM * SDIM * HDIM * 2);
  unsigned short* UwB  = (unsigned short*)alloc((size_t)HDIM * HDIM * 2);
  unsigned short* WwB  = (unsigned short*)alloc((size_t)HDIM * HDIM * 2);
  unsigned short* WihB = (unsigned short*)alloc((size_t)768 * 512 * 2);
  unsigned short* WhhB = (unsigned short*)alloc((size_t)768 * 256 * 2);
  unsigned short* fcwB = (unsigned short*)alloc((size_t)VDIM * HDIM * 2);
  unsigned short* embB = (unsigned short*)alloc((size_t)TDIM * BDIM * HDIM * 2);
  float* Ukeys = (float*)alloc((size_t)BDIM * SDIM * HDIM * 4);
  float* hbuf  = (float*)alloc((size_t)BDIM * HDIM * 4);
  unsigned short* hB = (unsigned short*)alloc((size_t)BDIM * HDIM * 2);
  float* qbuf = (float*)alloc((size_t)BDIM * HDIM * 4);
  unsigned short* xbuf = (unsigned short*)alloc((size_t)BDIM * 512 * 2);
  float* gi = (float*)alloc((size_t)BDIM * 768 * 4);
  float* gh = (float*)alloc((size_t)BDIM * 768 * 4);

  auto cvt = [&](const float* src, unsigned short* dst, int n) {
    cvt_f32_to_bf16<<<(n + 255) / 256, 256, 0, stream>>>(src, dst, n);
  };
  cvt(enc, encB, BDIM * SDIM * HDIM);
  cvt(Uw, UwB, HDIM * HDIM);
  cvt(Ww, WwB, HDIM * HDIM);
  cvt(W_ih, WihB, 768 * 512);
  cvt(W_hh, WhhB, 768 * 256);
  cvt(fc_w, fcwB, VDIM * HDIM);
  embed_kernel<<<(TDIM * BDIM * HDIM + 255) / 256, 256, 0, stream>>>(target, emb, embB);
  init_h_kernel<<<(BDIM * HDIM + 255) / 256, 256, 0, stream>>>(ehid, hbuf, hB, BDIM * HDIM);

  // K = 256 variant (waves = (M/64)*(N/32); % 8 == 0 at every call site)
  auto gemm256 = [&](const unsigned short* A, const unsigned short* W,
                     const float* bias, float* C, long long ldc, int M, int N) {
    int nPairs = N / 32;
    int blocks = (M / 64) * nPairs / 8;
    gemm_bf16_kernel<256><<<blocks, 256, 0, stream>>>(A, W, bias, C, ldc, nPairs);
  };
  // K = 512 variant
  auto gemm512 = [&](const unsigned short* A, const unsigned short* W,
                     const float* bias, float* C, long long ldc, int M, int N) {
    int nPairs = N / 32;
    int blocks = (M / 64) * nPairs / 8;
    gemm_bf16_kernel<512><<<blocks, 256, 0, stream>>>(A, W, bias, C, ldc, nPairs);
  };

  // Ukeys[b,s,:] = enc[b,s,:] @ Uw^T + Ub   (M = B*S = 3072)
  gemm256(encB, UwB, Ub, Ukeys, HDIM, BDIM * SDIM, HDIM);

  for (int t = 0; t < TDIM; ++t) {
    // q = h @ Ww^T + Wb
    gemm256(hB, WwB, Wb, qbuf, HDIM, BDIM, HDIM);
    // attention + build x = [emb_t | ctx] (bf16)
    attn_kernel<<<BDIM, 256, 0, stream>>>(qbuf, Ukeys, Vw, Vb, enc,
                                          embB + (size_t)t * BDIM * HDIM,
                                          xbuf, out_attn + t * SDIM);
    // gi = x @ W_ih^T + b_ih ; gh = h @ W_hh^T + b_hh
    gemm512(xbuf, WihB, b_ih, gi, 768, BDIM, 768);
    gemm256(hB, WhhB, b_hh, gh, 768, BDIM, 768);
    // GRU update
    gru_kernel<<<BDIM, HDIM, 0, stream>>>(gi, gh, hbuf, hB,
                                          (t == TDIM - 1) ? out_hlast : nullptr);
    // logits -> d_out rows [b, t, :], then in-place log_softmax
    gemm256(hB, fcwB, fc_b, out_logp + (long long)t * VDIM,
            (long long)TDIM * VDIM, BDIM, VDIM);
    logsoftmax_kernel<<<BDIM, 256, 0, stream>>>(out_logp + (long long)t * VDIM,
                                                (long long)TDIM * VDIM);
  }
}
